// Self_NonLocalBlock_90443421319299
// MI455X (gfx1250) — matrix operand order repaired and checked by reading, not compiled, not run on hardware
//
#include <hip/hip_runtime.h>
#include <hip/hip_bf16.h>
#include <math.h>

// ---------------------------------------------------------------------------
// Non-local self-attention block for MI455X (gfx1250, wave32, WMMA).
// Pipeline: convert/transpose -> QKV bf16 WMMA GEMM -> energy WMMA GEMM ->
// in-place row softmax -> (attn x v^T) WMMA GEMM + residual epilogue.
// Each GEMM wave computes a 32x64 f32 tile (8 WMMAs / K-step, 1.5 loads/WMMA).
// ---------------------------------------------------------------------------

typedef __bf16 bf16;
typedef bf16  v16bf __attribute__((ext_vector_type(16)));
typedef bf16  v8bf  __attribute__((ext_vector_type(8)));
typedef float v8f   __attribute__((ext_vector_type(8)));

#define NB 8
#define NC 512
#define NL 2048

// Build a 16-element bf16 fragment from two contiguous 8-element (16B) runs.
__device__ __forceinline__ v16bf load_frag(const bf16* lo, const bf16* hi) {
  v8bf a = *(const v8bf*)lo;
  v8bf b = *(const v8bf*)hi;
  v16bf r;
#pragma unroll
  for (int i = 0; i < 8; ++i) { r[i] = a[i]; r[i + 8] = b[i]; }
  return r;
}

// A-fragment (16x32 bf16, MxK), source row-major [M][K] with leading dim lda.
// Lane layout (ISA 7.12.2): m = lane&15, kg = lane>>4; runs at K=kg*8 and 16+kg*8.
__device__ __forceinline__ v16bf load_A(const bf16* src, int lda, int row0, int k0, int lane) {
  const bf16* base = src + (size_t)(row0 + (lane & 15)) * lda + k0 + ((lane >> 4) << 3);
  return load_frag(base, base + 16);
}

// B-fragment (32x16 bf16, KxN), source stored N-major: src[n*ldb + k] (K contiguous).
// Lane layout: n = lane&15, kh = lane>>4; runs at K = kh*8 and 16+kh*8.
__device__ __forceinline__ v16bf load_B(const bf16* src, int ldb, int n0, int k0, int lane) {
  const bf16* base = src + (size_t)(n0 + (lane & 15)) * ldb + k0 + ((lane >> 4) << 3);
  return load_frag(base, base + 16);
}

// --------------------------- kernel 0a: x -> xT bf16 ------------------------
__global__ void convert_transpose_x(const float* __restrict__ x, bf16* __restrict__ xT) {
  __shared__ float tile[32][33];
  const int b = blockIdx.x, c0 = blockIdx.y * 32, l0 = blockIdx.z * 32;
  const int tx = threadIdx.x, ty = threadIdx.y;   // blockDim (32,8)
  const float* xb = x + (size_t)b * NC * NL;
#pragma unroll
  for (int i = 0; i < 4; ++i)
    tile[ty + i * 8][tx] = xb[(size_t)(c0 + ty + i * 8) * NL + l0 + tx];
  __syncthreads();
  bf16* xtb = xT + (size_t)b * NL * NC;
#pragma unroll
  for (int i = 0; i < 4; ++i)
    xtb[(size_t)(l0 + ty + i * 8) * NC + c0 + tx] = (bf16)tile[tx][ty + i * 8];
}

// --------------------------- kernel 0b: weights -> bf16 ---------------------
__global__ void convert_w(const float* __restrict__ wq, const float* __restrict__ wk,
                          const float* __restrict__ wv, bf16* __restrict__ wout) {
  int i = blockIdx.x * blockDim.x + threadIdx.x;     // 3*NC*NC threads
  int which = i / (NC * NC), j = i % (NC * NC);
  const float* src = (which == 0) ? wq : (which == 1) ? wk : wv;
  wout[(size_t)which * NC * NC + j] = (bf16)src[j];
}

// --------------------------- kernel 1: QKV projection -----------------------
// qT[l,c] = sum_c' xT[l,c'] * W[c,c'] + bias[c]   (W row-major is exactly N-major B src)
// q,k stored [L,C]; v stored [C,L] (so it is K-contiguous as B operand later).
__global__ void __launch_bounds__(256) qkv_gemm(const bf16* __restrict__ xT,
                                                const bf16* __restrict__ wbf,
                                                const float* __restrict__ bq,
                                                const float* __restrict__ bk,
                                                const float* __restrict__ bv,
                                                bf16* __restrict__ qT,
                                                bf16* __restrict__ kT,
                                                bf16* __restrict__ vbuf) {
  const int lane = threadIdx.x & 31;
  int s = blockIdx.x * 8 + (threadIdx.x >> 5);       // 32x64 tile per wave
  const int nCt = NC / 64, nLt = NL / 32;
  const int which = s / (NB * nLt * nCt);
  int r = s % (NB * nLt * nCt);
  const int b = r / (nLt * nCt); r %= nLt * nCt;
  const int lt = r / nCt, ct = r % nCt;

  const bf16* A = xT + (size_t)b * NL * NC;
  const bf16* W = wbf + (size_t)which * NC * NC;
  const float* bias = (which == 0) ? bq : (which == 1) ? bk : bv;

  const int l0 = lt * 32, c0 = ct * 64;
  v8f acc[2][4] = {};
  for (int kk = 0; kk < NC; kk += 32) {
    v16bf a0 = load_A(A, NC, l0, kk, lane);
    v16bf a1 = load_A(A, NC, l0 + 16, kk, lane);
#pragma unroll
    for (int j = 0; j < 4; ++j) {
      v16bf bb = load_B(W, NC, c0 + j * 16, kk, lane);
      acc[0][j] = __builtin_amdgcn_wmma_f32_16x16x32_bf16(false, a0, false, bb,
                                                          (short)0, acc[0][j], false, false);
      acc[1][j] = __builtin_amdgcn_wmma_f32_16x16x32_bf16(false, a1, false, bb,
                                                          (short)0, acc[1][j], false, false);
    }
  }
  const int m = lane & 15, hg = lane >> 4;
#pragma unroll
  for (int i = 0; i < 2; ++i)
#pragma unroll
    for (int j = 0; j < 4; ++j) {
      const int c = c0 + j * 16 + m;
      const float bs = bias[c];
#pragma unroll
      for (int ri = 0; ri < 8; ++ri) {
        const int l = l0 + i * 16 + ri + hg * 8;
        const float v = acc[i][j][ri] + bs;
        if (which == 0)      qT[(size_t)b * NL * NC + (size_t)l * NC + c] = (bf16)v;
        else if (which == 1) kT[(size_t)b * NL * NC + (size_t)l * NC + c] = (bf16)v;
        else                 vbuf[(size_t)b * NC * NL + (size_t)c * NL + l] = (bf16)v;
      }
    }
}

// --------------------------- kernel 2: energy = q^T k -----------------------
__global__ void __launch_bounds__(256) energy_gemm(const bf16* __restrict__ qT,
                                                   const bf16* __restrict__ kT,
                                                   float* __restrict__ E) {
  const int lane = threadIdx.x & 31;
  int s = blockIdx.x * 8 + (threadIdx.x >> 5);
  const int nMt = NL / 64, nLt = NL / 32;
  const int b = s / (nLt * nMt);
  int r = s % (nLt * nMt);
  const int lt = r / nMt, mt = r % nMt;

  const bf16* A  = qT + (size_t)b * NL * NC;
  const bf16* Bk = kT + (size_t)b * NL * NC;
  const int l0 = lt * 32, m0 = mt * 64;
  v8f acc[2][4] = {};
  for (int kk = 0; kk < NC; kk += 32) {
    v16bf a0 = load_A(A, NC, l0, kk, lane);
    v16bf a1 = load_A(A, NC, l0 + 16, kk, lane);
#pragma unroll
    for (int j = 0; j < 4; ++j) {
      v16bf bb = load_B(Bk, NC, m0 + j * 16, kk, lane);
      acc[0][j] = __builtin_amdgcn_wmma_f32_16x16x32_bf16(false, a0, false, bb,
                                                          (short)0, acc[0][j], false, false);
      acc[1][j] = __builtin_amdgcn_wmma_f32_16x16x32_bf16(false, a1, false, bb,
                                                          (short)0, acc[1][j], false, false);
    }
  }
  float* Eb = E + (size_t)b * NL * NL;
  const int m = lane & 15, hg = lane >> 4;
#pragma unroll
  for (int i = 0; i < 2; ++i)
#pragma unroll
    for (int j = 0; j < 4; ++j)
#pragma unroll
      for (int ri = 0; ri < 8; ++ri)
        Eb[(size_t)(l0 + i * 16 + ri + hg * 8) * NL + m0 + j * 16 + m] = acc[i][j][ri];
}

// --------------------------- kernel 3: row softmax (in place) ---------------
// Reads a 2048-fp32 row, writes bf16 attn over the first half of the same row
// (bf16 row stride = 2*NL elements). Safe: values held in registers across sync.
__global__ void __launch_bounds__(256) softmax_rows(float* __restrict__ E) {
  __shared__ float red[256];
  float* p = E + (size_t)blockIdx.x * NL;
  const int t = threadIdx.x;
  float v[8];
#pragma unroll
  for (int j = 0; j < 8; ++j) v[j] = p[t + j * 256];
  float mx = -3.4e38f;
#pragma unroll
  for (int j = 0; j < 8; ++j) mx = fmaxf(mx, v[j]);
  red[t] = mx; __syncthreads();
  for (int sdist = 128; sdist > 0; sdist >>= 1) {
    if (t < sdist) red[t] = fmaxf(red[t], red[t + sdist]);
    __syncthreads();
  }
  mx = red[0]; __syncthreads();
  float sum = 0.f;
#pragma unroll
  for (int j = 0; j < 8; ++j) { v[j] = __expf(v[j] - mx); sum += v[j]; }
  red[t] = sum; __syncthreads();
  for (int sdist = 128; sdist > 0; sdist >>= 1) {
    if (t < sdist) red[t] += red[t + sdist];
    __syncthreads();
  }
  const float inv = 1.0f / red[0];
  bf16* q = (bf16*)p;
#pragma unroll
  for (int j = 0; j < 8; ++j) q[t + j * 256] = (bf16)(v[j] * inv);
}

// --------------------------- kernel 4: out = id + g*(attn x v^T)^T ----------
__global__ void __launch_bounds__(256) va_gemm(const float* __restrict__ Ebase,
                                               const bf16* __restrict__ vbuf,
                                               const float* __restrict__ idf,
                                               const float* __restrict__ gamma,
                                               float* __restrict__ out) {
  const int lane = threadIdx.x & 31;
  int s = blockIdx.x * 8 + (threadIdx.x >> 5);
  const int nCt = NC / 64, nLt = NL / 32;
  const int b = s / (nLt * nCt);
  int r = s % (nLt * nCt);
  const int lt = r / nCt, ct = r % nCt;

  const bf16* attn = (const bf16*)(Ebase + (size_t)b * NL * NL);  // row stride 2*NL bf16
  const bf16* Vb = vbuf + (size_t)b * NC * NL;
  const int l0 = lt * 32, c0 = ct * 64;
  v8f acc[2][4] = {};
  for (int kk = 0; kk < NL; kk += 32) {
    v16bf a0 = load_A(attn, 2 * NL, l0, kk, lane);
    v16bf a1 = load_A(attn, 2 * NL, l0 + 16, kk, lane);
#pragma unroll
    for (int j = 0; j < 4; ++j) {
      v16bf bb = load_B(Vb, NL, c0 + j * 16, kk, lane);
      acc[0][j] = __builtin_amdgcn_wmma_f32_16x16x32_bf16(false, a0, false, bb,
                                                          (short)0, acc[0][j], false, false);
      acc[1][j] = __builtin_amdgcn_wmma_f32_16x16x32_bf16(false, a1, false, bb,
                                                          (short)0, acc[1][j], false, false);
    }
  }
  const float g = gamma[0];
  const float* xb = idf + (size_t)b * NC * NL;
  float* ob = out + (size_t)b * NC * NL;
  const int m = lane & 15, hg = lane >> 4;
#pragma unroll
  for (int i = 0; i < 2; ++i)
#pragma unroll
    for (int j = 0; j < 4; ++j)
#pragma unroll
      for (int ri = 0; ri < 8; ++ri) {
        const size_t idx = (size_t)(c0 + j * 16 + m) * NL + l0 + i * 16 + ri + hg * 8;
        ob[idx] = xb[idx] + g * acc[i][j][ri];
      }
}

// ---------------------------------------------------------------------------
extern "C" void kernel_launch(void* const* d_in, const int* in_sizes, int n_in,
                              void* d_out, int out_size, void* d_ws, size_t ws_size,
                              hipStream_t stream) {
  (void)in_sizes; (void)n_in; (void)out_size; (void)ws_size;
  const float* idf   = (const float*)d_in[0];
  const float* Wq    = (const float*)d_in[1];
  const float* bq    = (const float*)d_in[2];
  const float* Wk    = (const float*)d_in[3];
  const float* bk    = (const float*)d_in[4];
  const float* Wv    = (const float*)d_in[5];
  const float* bv    = (const float*)d_in[6];
  const float* gamma = (const float*)d_in[7];
  float* out = (float*)d_out;

  char* ws = (char*)d_ws;
  const size_t E_bytes = (size_t)NB * NL * NL * 4;      // 128 MB energy / attn (in place)
  const size_t tbytes  = (size_t)NB * NL * NC * 2;      // 16 MB per q/k/v tensor
  bf16*  xT  = (bf16*)ws;                                // overlaps E (dead before K2)
  float* E   = (float*)ws;
  bf16*  qT  = (bf16*)(ws + E_bytes);
  bf16*  kT  = (bf16*)(ws + E_bytes + tbytes);
  bf16*  vb  = (bf16*)(ws + E_bytes + 2 * tbytes);
  bf16*  wbf = (bf16*)(ws + E_bytes + 3 * tbytes);

  convert_transpose_x<<<dim3(NB, NC / 32, NL / 32), dim3(32, 8), 0, stream>>>(idf, xT);
  convert_w<<<(3 * NC * NC) / 256, 256, 0, stream>>>(Wq, Wk, Wv, wbf);
  qkv_gemm<<<(3 * NB * (NL / 32) * (NC / 64)) / 8, 256, 0, stream>>>(xT, wbf, bq, bk, bv, qT, kT, vb);
  energy_gemm<<<(NB * (NL / 32) * (NL / 64)) / 8, 256, 0, stream>>>(qT, kT, E);
  softmax_rows<<<NB * NL, 256, 0, stream>>>(E);
  va_gemm<<<(NB * (NL / 32) * (NC / 64)) / 8, 256, 0, stream>>>(E, vb, idf, gamma, out);
}
